// NodeGCN_3985729651396
// MI455X (gfx1250) — matrix operand order repaired and checked by reading, not compiled, not run on hardware
//
#include <hip/hip_runtime.h>
#include <hip/hip_bf16.h>

typedef __attribute__((ext_vector_type(16))) __bf16 v16bf;
typedef __attribute__((ext_vector_type(8)))  float  v8f;

#define NNODES 100000
#define NEDGES 1600000
#define FIN 256
#define HIDN 256
#define NCLS 40

__global__ void fill_kernel(float* p, float v, long long n) {
  long long i = (long long)blockIdx.x * blockDim.x + threadIdx.x;
  if (i < n) p[i] = v;
}

__global__ void deg_acc_kernel(const long long* __restrict__ dst, float* deg, int E) {
  int i = blockIdx.x * blockDim.x + threadIdx.x;
  if (i < E) unsafeAtomicAdd(&deg[dst[i]], 1.0f);
}

__global__ void rsqrt_kernel(float* d, int n) {
  int i = blockIdx.x * blockDim.x + threadIdx.x;
  if (i < n) d[i] = rsqrtf(d[i]);
}

// Pack W[K x M] (f32) into WMMA B-fragment order (bf16), zero-padding ragged M.
// Packed index: ((tm*(K/32) + kb)*32 + lane)*16 + e
//   lane half h=lane>>4, n=lane&15 ; element e -> B[kb*32 + h*16 + e][tm*16 + n]
__global__ void pack_w_kernel(const float* __restrict__ W, __bf16* __restrict__ Wp,
                              int K, int M, int tiles_m) {
  int idx = blockIdx.x * blockDim.x + threadIdx.x;
  int total = tiles_m * (K >> 5) * 512;
  if (idx >= total) return;
  int e    = idx & 15;
  int lane = (idx >> 4) & 31;
  int kb   = (idx >> 9) % (K >> 5);
  int tm   = (idx >> 9) / (K >> 5);
  int k    = kb * 32 + (lane >> 4) * 16 + e;
  int col  = tm * 16 + (lane & 15);
  Wp[idx] = (col < M) ? (__bf16)W[k * M + col] : (__bf16)0.0f;
}

// C[N x M] = A[N x K] (f32 -> bf16 on load) @ W (pre-packed bf16 fragments), f32 acc.
// N % 16 == 0, K % 32 == 0. Each wave computes CT consecutive 16x16 col tiles:
// per K-step, one shared A fragment, ALL CT B fragments loaded in one clause,
// then CT back-to-back WMMAs (one load-wait per K-step instead of CT).
template <int CT>
__global__ void gemm_wmma_bf16(const float* __restrict__ A, const __bf16* __restrict__ Wp,
                               float* __restrict__ C, int N, int K, int M) {
  const int lane    = threadIdx.x & 31;
  const int wave    = threadIdx.x >> 5;
  const int tiles_m = (M + 15) >> 4;
  const int groups  = (tiles_m + CT - 1) / CT;
  const int kb_n    = K >> 5;
  const int t = blockIdx.x * (blockDim.x >> 5) + wave;
  if (t >= (N >> 4) * groups) return;            // wave-uniform: EXEC stays all-1s
  const int tn = t / groups, g = t % groups;
  const int row0 = tn << 4;
  const int m = lane & 15;                       // A row in tile / B,C column in tile
  const int h = lane >> 4;                       // lane half

  v8f acc[CT];
#pragma unroll
  for (int j = 0; j < CT; ++j) acc[j] = (v8f){};

  const float* arow = A + (long long)(row0 + m) * K;
  for (int kb = 0; kb < kb_n; ++kb) {
    const int k0 = kb << 5;
    v16bf a;
#pragma unroll
    for (int e = 0; e < 8; ++e) a[e]     = (__bf16)arow[k0 + h * 16 + e];
#pragma unroll
    for (int e = 0; e < 8; ++e) a[8 + e] = (__bf16)arow[k0 + h * 16 + 8 + e];
    v16bf b[CT];
#pragma unroll
    for (int j = 0; j < CT; ++j)
      b[j] = *(const v16bf*)(Wp +
          ((((long long)(g * CT + j) * kb_n + kb) << 9) + (lane << 4)));
#pragma unroll
    for (int j = 0; j < CT; ++j)
      acc[j] = __builtin_amdgcn_wmma_f32_16x16x32_bf16(false, a, false, b[j],
                                                       (short)0, acc[j], false, false);
  }
#pragma unroll
  for (int j = 0; j < CT; ++j) {
    const int col = ((g * CT + j) << 4) + m;
    if (col < M) {
#pragma unroll
      for (int r = 0; r < 8; ++r)
        C[(long long)(row0 + r + 8 * h) * M + col] = acc[j][r];
    }
  }
}

// One wave per edge (E real edges + N self-loops). out[dst] += dinv[src]*dinv[dst]*H[src]
__global__ void scatter_kernel(const float* __restrict__ H, float* __restrict__ out,
                               const long long* __restrict__ src,
                               const long long* __restrict__ dst,
                               const float* __restrict__ dinv, int E, int Nn, int F) {
  int w = (int)((blockIdx.x * blockDim.x + threadIdx.x) >> 5);
  int lane = threadIdx.x & 31;
  if (w >= E + Nn) return;
  long long s, d;
  if (w < E) { s = src[w]; d = dst[w]; }
  else       { s = d = (long long)(w - E); }
  float nrm = dinv[s] * dinv[d];
  const float* hs = H + s * (long long)F;
  float* od = out + d * (long long)F;
  for (int f = lane; f < F; f += 32)
    unsafeAtomicAdd(&od[f], nrm * hs[f]);
}

__global__ void bias_act_kernel(float* x, const float* __restrict__ b,
                                long long n, int F, int relu) {
  long long i = (long long)blockIdx.x * blockDim.x + threadIdx.x;
  if (i < n) {
    float v = x[i] + b[(int)(i % F)];
    x[i] = (relu && v < 0.0f) ? 0.0f : v;
  }
}

extern "C" void kernel_launch(void* const* d_in, const int* in_sizes, int n_in,
                              void* d_out, int out_size, void* d_ws, size_t ws_size,
                              hipStream_t stream) {
  const float*     x  = (const float*)d_in[0];
  const long long* ei = (const long long*)d_in[1];   // int64 edge_index [2, E]
  const float*     W1 = (const float*)d_in[2];
  const float*     b1 = (const float*)d_in[3];
  const float*     W2 = (const float*)d_in[4];
  const float*     b2 = (const float*)d_in[5];
  const float*     W3 = (const float*)d_in[6];
  const float*     b3 = (const float*)d_in[7];
  float* out = (float*)d_out;

  const int N = NNODES, E = NEDGES;
  const long long NH = (long long)N * HIDN;
  const long long NC = (long long)N * NCLS;

  // workspace layout
  float*  dinv = (float*)d_ws;          // N
  float*  H    = dinv + N;              // N*HID
  float*  AGG  = H + NH;                // N*HID
  __bf16* Wp   = (__bf16*)(AGG + NH);   // packed weights, up to 256*256 bf16

  const long long* srcp = ei;
  const long long* dstp = ei + E;

  // ---- symmetric GCN normalization: deg includes self-loop (init 1.0)
  fill_kernel<<<(N + 255) / 256, 256, 0, stream>>>(dinv, 1.0f, (long long)N);
  deg_acc_kernel<<<(E + 255) / 256, 256, 0, stream>>>(dstp, dinv, E);
  rsqrt_kernel<<<(N + 255) / 256, 256, 0, stream>>>(dinv, N);

  const int totalE = E + N;
  const int scat_blocks = (totalE + 7) / 8;         // 8 edge-waves per 256-thread block

  // GEMM launch geometry: 4 waves per 128-thread block
  const int waves_256 = (N / 16) * ((HIDN / 16) / 4);   // CT=4 -> 4 col-tile groups
  const int gemm_blocks_256 = (waves_256 + 3) / 4;
  const int waves_40 = (N / 16) * 1;                    // CT=3 -> 1 group (tiles_m=3)
  const int gemm_blocks_40 = (waves_40 + 3) / 4;
  const int packn_256 = 16 * (256 / 32) * 512;
  const int packn_40  = 3  * (256 / 32) * 512;

  // ---- layer 1: H = x@W1 ; AGG = scatter(H) ; AGG = relu(AGG + b1)
  pack_w_kernel<<<(packn_256 + 255) / 256, 256, 0, stream>>>(W1, Wp, FIN, HIDN, 16);
  gemm_wmma_bf16<4><<<gemm_blocks_256, 128, 0, stream>>>(x, Wp, H, N, FIN, HIDN);
  fill_kernel<<<(int)((NH + 255) / 256), 256, 0, stream>>>(AGG, 0.0f, NH);
  scatter_kernel<<<scat_blocks, 256, 0, stream>>>(H, AGG, srcp, dstp, dinv, E, N, HIDN);
  bias_act_kernel<<<(int)((NH + 255) / 256), 256, 0, stream>>>(AGG, b1, NH, HIDN, 1);

  // ---- layer 2: H = AGG@W2 ; AGG = scatter(H) ; AGG = relu(AGG + b2)
  pack_w_kernel<<<(packn_256 + 255) / 256, 256, 0, stream>>>(W2, Wp, HIDN, HIDN, 16);
  gemm_wmma_bf16<4><<<gemm_blocks_256, 128, 0, stream>>>(AGG, Wp, H, N, HIDN, HIDN);
  fill_kernel<<<(int)((NH + 255) / 256), 256, 0, stream>>>(AGG, 0.0f, NH);
  scatter_kernel<<<scat_blocks, 256, 0, stream>>>(H, AGG, srcp, dstp, dinv, E, N, HIDN);
  bias_act_kernel<<<(int)((NH + 255) / 256), 256, 0, stream>>>(AGG, b2, NH, HIDN, 1);

  // ---- layer 3: H = AGG@W3 (N x 40) ; out = scatter(H) + b3 (no relu)
  pack_w_kernel<<<(packn_40 + 255) / 256, 256, 0, stream>>>(W3, Wp, HIDN, NCLS, 3);
  gemm_wmma_bf16<3><<<gemm_blocks_40, 128, 0, stream>>>(AGG, Wp, H, N, HIDN, NCLS);
  fill_kernel<<<(int)((NC + 255) / 256), 256, 0, stream>>>(out, 0.0f, NC);
  scatter_kernel<<<scat_blocks, 256, 0, stream>>>(H, out, srcp, dstp, dinv, E, N, NCLS);
  bias_act_kernel<<<(int)((NC + 255) / 256), 256, 0, stream>>>(out, b3, NC, NCLS, 0);
}
